// DelayCell_81707457839184
// MI455X (gfx1250) — matrix operand order repaired and checked by reading, not compiled, not run on hardware
//
#include <hip/hip_runtime.h>
#include <hip/hip_bf16.h>
#include <math.h>

typedef __attribute__((ext_vector_type(16))) _Float16 v16h;
typedef __attribute__((ext_vector_type(8)))  float    v8f;

#define UNITS  256
#define ORDER  6
#define IDIM   256
#define BATCH  32
#define SEQ    2048
#define SEGS   32
#define SEGLEN 64   // SEQ / SEGS

// workspace layout (float element offsets)
#define OFF_M   ((size_t)0)                         // 256*36  per-unit M = I + AT/theta
#define OFF_P   (OFF_M + (size_t)UNITS * 36)        // 256*36  per-unit P = M^SEGLEN
#define OFF_C   (OFF_P + (size_t)UNITS * 36)        // 256*6   per-unit decoder vector
#define OFF_BI  (OFF_C + (size_t)UNITS * ORDER)     // 256*6   per-unit B/theta
#define OFF_D   (OFF_BI + (size_t)UNITS * ORDER)    // B*U*S*6 segment drive vectors
#define OFF_XS  (OFF_D + (size_t)BATCH * UNITS * SEGS * ORDER) // B*U*S*6 segment start states

// ---------------------------------------------------------------------------
// K1: U[bt, j] = inputs[bt, :] @ encoders[:, j]   via v_wmma_f32_16x16x32_f16
// grid = (16, 512), block = 256 (8 waves). blockIdx.x -> 16-col slice of
// encoders (staged to LDS as f16 in B-fragment-friendly layout), each wave
// owns one 16-row M tile, K=256 in 8 steps of 32.
// ---------------------------------------------------------------------------
__global__ __launch_bounds__(256)
void k1_encode_gemm(const float* __restrict__ inp,   // [B*T, 256]
                    const float* __restrict__ enc,   // [256, 256]
                    float* __restrict__ U)           // [B*T, 256]  (== d_out)
{
    __shared__ _Float16 encB[16 * 264];              // [n][k] padded, 8448 B
    const int j0 = blockIdx.x * 16;
    {   // stage: encB[n*264 + d] = (f16) enc[d*256 + j0 + n]
        const int d = threadIdx.x;                   // 0..255
        const float4* e4 = (const float4*)(enc + (size_t)d * UNITS + j0);
        float4 q0 = e4[0], q1 = e4[1], q2 = e4[2], q3 = e4[3];
        float v[16] = {q0.x,q0.y,q0.z,q0.w, q1.x,q1.y,q1.z,q1.w,
                       q2.x,q2.y,q2.z,q2.w, q3.x,q3.y,q3.z,q3.w};
        #pragma unroll
        for (int n = 0; n < 16; ++n) encB[n * 264 + d] = (_Float16)v[n];
    }
    __syncthreads();

    const int lane = threadIdx.x & 31;
    const int wave = threadIdx.x >> 5;
    const int col  = lane & 15;                      // N (and M-row for A)
    const int grp  = lane >> 4;                      // lane group 0/1
    const int m0   = (blockIdx.y * 8 + wave) * 16;
    const float* arow = inp + (size_t)(m0 + col) * IDIM;

    v8f acc = {};
    #pragma unroll
    for (int s = 0; s < 8; ++s) {
        const int kb = s * 32;
        // A fragment: halves 0..7 -> K = kb + 8*grp + h ; 8..15 -> K = kb+16+8*grp + (h-8)
        v16h a;
        {
            const float4* p0 = (const float4*)(arow + kb + 8 * grp);
            const float4* p1 = (const float4*)(arow + kb + 16 + 8 * grp);
            float4 f0 = p0[0], f1 = p0[1], f2 = p1[0], f3 = p1[1];
            a[0]=(_Float16)f0.x; a[1]=(_Float16)f0.y; a[2]=(_Float16)f0.z; a[3]=(_Float16)f0.w;
            a[4]=(_Float16)f1.x; a[5]=(_Float16)f1.y; a[6]=(_Float16)f1.z; a[7]=(_Float16)f1.w;
            a[8]=(_Float16)f2.x; a[9]=(_Float16)f2.y; a[10]=(_Float16)f2.z; a[11]=(_Float16)f2.w;
            a[12]=(_Float16)f3.x; a[13]=(_Float16)f3.y; a[14]=(_Float16)f3.z; a[15]=(_Float16)f3.w;
        }
        // B fragment: lane L: N = L%16, halves 0..7 -> K = kb + 8*grp + h ; 8..15 -> K = kb+16+8*grp + (h-8)
        v16h b;
        {
            const _Float16* bp = &encB[col * 264 + kb + 8 * grp];
            #pragma unroll
            for (int h = 0; h < 16; ++h) b[h] = bp[h < 8 ? h : h + 8];
        }
        acc = __builtin_amdgcn_wmma_f32_16x16x32_f16(false, a, false, b,
                                                     (short)0, acc, false, false);
    }
    // D layout: VGPR r -> M = 8*grp + r, N = col
    float* urow = U + (size_t)(m0 + 8 * grp) * UNITS + j0 + col;
    #pragma unroll
    for (int r = 0; r < 8; ++r) urow[(size_t)r * UNITS] = acc[r];
}

// ---------------------------------------------------------------------------
// K2: per-unit constants. 1 block of 256 threads, thread u:
//   M = I + AT/theta_u ; P = M^SEGLEN ; bi = Bmat/theta_u ; C from decoders.
// ---------------------------------------------------------------------------
__global__ __launch_bounds__(256)
void k2_setup(const float* __restrict__ theta, const float* __restrict__ AT,
              const float* __restrict__ Bm,    const float* __restrict__ dec,
              float* __restrict__ ws)
{
    const int u = threadIdx.x;
    const float inv = 1.0f / theta[u];
    float M[36], P[36], T[36];
    #pragma unroll
    for (int o = 0; o < 6; ++o)
        #pragma unroll
        for (int p = 0; p < 6; ++p)
            M[o * 6 + p] = ((o == p) ? 1.0f : 0.0f) + AT[o * 6 + p] * inv;
    #pragma unroll
    for (int i = 0; i < 36; ++i) P[i] = M[i];
    for (int sq = 0; sq < 6; ++sq) {                 // P = M^64
        #pragma unroll
        for (int o = 0; o < 6; ++o)
            #pragma unroll
            for (int p = 0; p < 6; ++p) {
                float s = 0.0f;
                #pragma unroll
                for (int k = 0; k < 6; ++k) s += P[o * 6 + k] * P[k * 6 + p];
                T[o * 6 + p] = s;
            }
        #pragma unroll
        for (int i = 0; i < 36; ++i) P[i] = T[i];
    }
    #pragma unroll
    for (int i = 0; i < 36; ++i) { ws[OFF_M + (size_t)u * 36 + i] = M[i];
                                   ws[OFF_P + (size_t)u * 36 + i] = P[i]; }
    #pragma unroll
    for (int p = 0; p < ORDER; ++p) ws[OFF_BI + (size_t)u * ORDER + p] = Bm[p] * inv;
    #pragma unroll
    for (int o = 0; o < ORDER; ++o)
        ws[OFF_C + (size_t)u * ORDER + o] = dec[(size_t)(u * ORDER + o) * UNITS + u];
}

// ---------------------------------------------------------------------------
// K3: per-segment drive vector (recurrence from zero state, no outputs).
// grid = BATCH*SEGS blocks, 256 threads (one per unit). Coalesced U reads.
// ---------------------------------------------------------------------------
__global__ __launch_bounds__(256)
void k3_segparts(const float* __restrict__ U, float* __restrict__ ws)
{
    const int b = blockIdx.x >> 5;                   // / SEGS
    const int s = blockIdx.x & (SEGS - 1);
    const int u = threadIdx.x;
    float M[36], bi[ORDER];
    #pragma unroll
    for (int i = 0; i < 36; ++i) M[i] = ws[OFF_M + (size_t)u * 36 + i];
    #pragma unroll
    for (int p = 0; p < ORDER; ++p) bi[p] = ws[OFF_BI + (size_t)u * ORDER + p];
    float x[ORDER] = {0,0,0,0,0,0};
    const float* up = U + ((size_t)b * SEQ + (size_t)s * SEGLEN) * UNITS + u;
    for (int i = 0; i < SEGLEN; ++i) {
        const float uv = up[(size_t)i * UNITS];
        float xn[ORDER];
        #pragma unroll
        for (int p = 0; p < ORDER; ++p) {
            float a = uv * bi[p];
            #pragma unroll
            for (int o = 0; o < ORDER; ++o) a += x[o] * M[o * 6 + p];
            xn[p] = a;
        }
        #pragma unroll
        for (int p = 0; p < ORDER; ++p) x[p] = xn[p];
    }
    float* dp = ws + OFF_D + (((size_t)b * UNITS + u) * SEGS + s) * ORDER;
    #pragma unroll
    for (int o = 0; o < ORDER; ++o) dp[o] = x[o];
}

// ---------------------------------------------------------------------------
// K4: sequential scan over segments (only 32 steps): xs[s+1] = xs[s]@P + d[s]
// grid = BATCH blocks, 256 threads (one per unit).
// ---------------------------------------------------------------------------
__global__ __launch_bounds__(256)
void k4_scan(const float* __restrict__ x0, float* __restrict__ ws)
{
    const int b = blockIdx.x;
    const int u = threadIdx.x;
    float P[36];
    #pragma unroll
    for (int i = 0; i < 36; ++i) P[i] = ws[OFF_P + (size_t)u * 36 + i];
    float x[ORDER];
    #pragma unroll
    for (int o = 0; o < ORDER; ++o)
        x[o] = x0[(size_t)b * (UNITS * ORDER) + u * ORDER + o];
    const size_t base = ((size_t)b * UNITS + u) * SEGS * ORDER;
    for (int s = 0; s < SEGS; ++s) {
        float* xs = ws + OFF_XS + base + (size_t)s * ORDER;
        #pragma unroll
        for (int o = 0; o < ORDER; ++o) xs[o] = x[o];
        const float* d = ws + OFF_D + base + (size_t)s * ORDER;
        float xn[ORDER];
        #pragma unroll
        for (int p = 0; p < ORDER; ++p) {
            float a = d[p];
            #pragma unroll
            for (int o = 0; o < ORDER; ++o) a += x[o] * P[o * 6 + p];
            xn[p] = a;
        }
        #pragma unroll
        for (int o = 0; o < ORDER; ++o) x[o] = xn[o];
    }
}

// ---------------------------------------------------------------------------
// K5: replay each segment from its start state, emit y = tanh(C.x) IN PLACE
// over the U buffer (== d_out). Each thread reads U[b,t,u] then overwrites
// the same element with y[b,t,u]; (b,s) blocks own disjoint t ranges.
// ---------------------------------------------------------------------------
__global__ __launch_bounds__(256)
void k5_emit(float* __restrict__ UY, const float* __restrict__ ws)
{
    const int b = blockIdx.x >> 5;
    const int s = blockIdx.x & (SEGS - 1);
    const int u = threadIdx.x;
    float M[36], bi[ORDER], cv[ORDER];
    #pragma unroll
    for (int i = 0; i < 36; ++i) M[i] = ws[OFF_M + (size_t)u * 36 + i];
    #pragma unroll
    for (int p = 0; p < ORDER; ++p) {
        bi[p] = ws[OFF_BI + (size_t)u * ORDER + p];
        cv[p] = ws[OFF_C  + (size_t)u * ORDER + p];
    }
    float x[ORDER];
    const size_t base = (((size_t)b * UNITS + u) * SEGS + s) * ORDER;
    #pragma unroll
    for (int o = 0; o < ORDER; ++o) x[o] = ws[OFF_XS + base + o];
    float* p = UY + ((size_t)b * SEQ + (size_t)s * SEGLEN) * UNITS + u;
    for (int i = 0; i < SEGLEN; ++i) {
        const float uv = p[(size_t)i * UNITS];
        float xn[ORDER];
        #pragma unroll
        for (int q = 0; q < ORDER; ++q) {
            float a = uv * bi[q];
            #pragma unroll
            for (int o = 0; o < ORDER; ++o) a += x[o] * M[o * 6 + q];
            xn[q] = a;
        }
        #pragma unroll
        for (int q = 0; q < ORDER; ++q) x[q] = xn[q];
        float y = 0.0f;
        #pragma unroll
        for (int o = 0; o < ORDER; ++o) y += cv[o] * x[o];
        p[(size_t)i * UNITS] = tanhf(y);
    }
}

extern "C" void kernel_launch(void* const* d_in, const int* in_sizes, int n_in,
                              void* d_out, int out_size, void* d_ws, size_t ws_size,
                              hipStream_t stream)
{
    (void)in_sizes; (void)n_in; (void)out_size; (void)ws_size;
    const float* inp   = (const float*)d_in[0];   // [32, 2048, 256]
    const float* x0    = (const float*)d_in[1];   // [32, 1536]
    const float* enc   = (const float*)d_in[2];   // [256, 256]
    const float* theta = (const float*)d_in[3];   // [1, 256, 1]
    const float* dec   = (const float*)d_in[4];   // [1536, 256]
    const float* AT    = (const float*)d_in[5];   // [6, 6]
    const float* Bm    = (const float*)d_in[6];   // [1, 1, 6]
    float* out = (float*)d_out;                   // [32, 2048, 256]
    float* ws  = (float*)d_ws;

    // K1: encoder GEMM straight into d_out (acts as U buffer, overwritten by K5)
    dim3 g1(16, (BATCH * SEQ) / (16 * 8));        // 16 x 512 blocks
    k1_encode_gemm<<<g1, 256, 0, stream>>>(inp, enc, out);
    k2_setup<<<1, 256, 0, stream>>>(theta, AT, Bm, dec, ws);
    k3_segparts<<<BATCH * SEGS, 256, 0, stream>>>(out, ws);
    k4_scan<<<BATCH, 256, 0, stream>>>(x0, ws);
    k5_emit<<<BATCH * SEGS, 256, 0, stream>>>(out, ws);
}
